// AttentionBlock_17085379903961
// MI455X (gfx1250) — compile-verified, matrix operand order checked by reading, not run on hardware
//
#include <hip/hip_runtime.h>

typedef __attribute__((ext_vector_type(16))) _Float16 v16h;
typedef __attribute__((ext_vector_type(8)))  _Float16 v8h;
typedef __attribute__((ext_vector_type(8)))  float    v8f;

#define QKSCALE 0.35355339059327373f  // 1/sqrt(sqrt(64))

// ---------------------------------------------------------------------------
// Fragment loader for V_WMMA_F32_16X16X32_F16, per CDNA5 ISA §7.12.2.
// A-matrix 16x32 (row-major, stride ld):
//   lanes 0-15  hold row M=lane,    K = {0..7, 16..23}
//   lanes 16-31 hold row M=lane-16, K = {8..15, 24..31}
// i.e. per lane: 8 contiguous halves at (row*ld + 8*half) and at (+16).
// Dense B (32x16) is the mirror image with lane = column, so loading the
// second operand from an N-major (transposed) buffer uses the same helper.
// ---------------------------------------------------------------------------
__device__ __forceinline__ v16h load_frag(const _Float16* __restrict__ p, int ld) {
  const int lane = threadIdx.x & 31;
  const int half = lane >> 4;
  const int row  = lane & 15;
  const _Float16* q = p + row * ld + (half << 3);
  v8h lo = *(const v8h*)(q);        // K = 8*half .. 8*half+7
  v8h hi = *(const v8h*)(q + 16);   // K = 16+8*half .. 16+8*half+7
  return __builtin_shufflevector(lo, hi, 0,1,2,3,4,5,6,7,8,9,10,11,12,13,14,15);
}

__device__ __forceinline__ v8f wmma16(v16h a, v16h b, v8f c) {
  return __builtin_amdgcn_wmma_f32_16x16x32_f16(false, a, false, b, (short)0, c,
                                                false, false);
}

// ---------------------------------------------------------------------------
// fp32 -> f16 weight conversion
// ---------------------------------------------------------------------------
__global__ void cvt_f16_kernel(const float* __restrict__ in,
                               _Float16* __restrict__ out, int n) {
  int i = blockIdx.x * blockDim.x + threadIdx.x;
  if (i < n) out[i] = (_Float16)in[i];
}

// ---------------------------------------------------------------------------
// GroupNorm over (16 ch x 1024) per (b,g); writes xn^T (b, L=1024, C=512) f16
// ---------------------------------------------------------------------------
__global__ void groupnorm_kernel(const float* __restrict__ x,
                                 const float* __restrict__ gw,
                                 const float* __restrict__ gb,
                                 _Float16* __restrict__ xnt) {
  const int C = 512, L = 1024, CG = 16, NE = CG * L;
  const int b = blockIdx.x >> 5;
  const int g = blockIdx.x & 31;
  const float* xp = x + ((size_t)b * C + g * CG) * L;

  float s = 0.f, ss = 0.f;
  for (int i = threadIdx.x; i < NE; i += blockDim.x) {
    float v = xp[i];           // l-fastest: coalesced
    s += v; ss += v * v;
  }
  const int lane = threadIdx.x & 31, wave = threadIdx.x >> 5;
  #pragma unroll
  for (int off = 16; off >= 1; off >>= 1) {
    s  += __shfl_xor(s,  off, 32);
    ss += __shfl_xor(ss, off, 32);
  }
  __shared__ float rs[8], rss[8];
  if (lane == 0) { rs[wave] = s; rss[wave] = ss; }
  __syncthreads();
  float tot = 0.f, tot2 = 0.f;
  #pragma unroll
  for (int i = 0; i < 8; i++) { tot += rs[i]; tot2 += rss[i]; }
  const float mean = tot * (1.f / NE);
  const float var  = tot2 * (1.f / NE) - mean * mean;
  const float inv  = rsqrtf(var + 1e-5f);

  // write c-fastest so stores to xn^T are contiguous in C
  for (int i = threadIdx.x; i < NE; i += blockDim.x) {
    int cl = i & 15;
    int l  = i >> 4;
    int c  = g * CG + cl;
    float v = xp[(size_t)cl * L + l];            // cached from pass 1
    float y = (v - mean) * inv * gw[c] + gb[c];
    xnt[((size_t)b * L + l) * C + c] = (_Float16)y;
  }
}

// ---------------------------------------------------------------------------
// QKV GEMM: per batch  C(1536 x 1024) = W(1536x512) * Xn(512x1024) + bias.
// NT form: A = W row-major, Bt = xn^T (L x C).  Epilogue scatters to
//   q,k: (b,h,L,64) t-major (scale folded in),  v: (b,h,64,L) c-major.
// Block = 8 waves, wave tile 32x32 (2x2 WMMA), block tile 128(M) x 64(N).
// ---------------------------------------------------------------------------
__global__ void qkv_gemm_kernel(const _Float16* __restrict__ Wh,
                                const _Float16* __restrict__ xnt,
                                const float* __restrict__ qkvb,
                                _Float16* __restrict__ qt,
                                _Float16* __restrict__ kt,
                                _Float16* __restrict__ vv) {
  const int C = 512, L = 1024;
  const int b    = blockIdx.z;
  const int wave = threadIdx.x >> 5, lane = threadIdx.x & 31;
  const int half = lane >> 4,        nloc = lane & 15;
  const int m0 = blockIdx.x * 128 + (wave & 3) * 32;
  const int n0 = blockIdx.y * 64  + (wave >> 2) * 32;
  const _Float16* Bt = xnt + (size_t)b * L * C;

  v8f c00 = {}, c01 = {}, c10 = {}, c11 = {};
  for (int k = 0; k < C; k += 32) {
    __builtin_prefetch(Wh + (size_t)m0 * C + k + 32, 0, 1);
    __builtin_prefetch(Bt + (size_t)n0 * C + k + 32, 0, 1);
    v16h a0 = load_frag(Wh + (size_t)m0 * C + k, C);
    v16h a1 = load_frag(Wh + (size_t)(m0 + 16) * C + k, C);
    v16h b0 = load_frag(Bt + (size_t)n0 * C + k, C);
    v16h b1 = load_frag(Bt + (size_t)(n0 + 16) * C + k, C);
    c00 = wmma16(a0, b0, c00);
    c01 = wmma16(a0, b1, c01);
    c10 = wmma16(a1, b0, c10);
    c11 = wmma16(a1, b1, c11);
  }

  auto store_tile = [&](v8f acc, int tm, int tn) {
    #pragma unroll
    for (int r = 0; r < 8; r++) {
      int o = m0 + tm * 16 + r + half * 8;       // C-layout: M = r + 8*half
      int l = n0 + tn * 16 + nloc;               //           N = lane&15
      float val = acc[r] + qkvb[o];
      int hh = o / 192, rem = o - hh * 192;
      int which = rem >> 6, ch = rem & 63;
      size_t bh = (size_t)(b * 8 + hh);
      if (which == 0)
        qt[(bh * L + l) * 64 + ch] = (_Float16)(val * QKSCALE);
      else if (which == 1)
        kt[(bh * L + l) * 64 + ch] = (_Float16)(val * QKSCALE);
      else
        vv[(bh * 64 + ch) * L + l] = (_Float16)val;
    }
  };
  store_tile(c00, 0, 0); store_tile(c01, 0, 1);
  store_tile(c10, 1, 0); store_tile(c11, 1, 1);
}

// ---------------------------------------------------------------------------
// Flash attention per (b,h): each wave owns 16 t-rows, loops s in steps of 32.
// S tiles via WMMA (K=64 = 2 steps), online softmax with __shfl_xor row
// reductions, P converted C-layout -> A-fragment via per-wave LDS tile,
// then 4 WMMAs against V (c-major).  Output written t-major into a^T (LxC).
// ---------------------------------------------------------------------------
__global__ void attn_kernel(const _Float16* __restrict__ qt,
                            const _Float16* __restrict__ kt,
                            const _Float16* __restrict__ vv,
                            _Float16* __restrict__ at) {
  const int L = 1024, HD = 64, C = 512;
  const int bh = blockIdx.y;                    // 0..63
  const int b = bh >> 3, h = bh & 7;
  const int wave = threadIdx.x >> 5, lane = threadIdx.x & 31;
  const int half = lane >> 4, nloc = lane & 15;
  const int t0 = blockIdx.x * 128 + wave * 16;

  const _Float16* q = qt + (size_t)bh * L * HD;
  const _Float16* k = kt + (size_t)bh * L * HD;
  const _Float16* v = vv + (size_t)bh * HD * L;

  __shared__ __align__(16) _Float16 pbuf[8][16 * 32];

  const v16h aq0 = load_frag(q + (size_t)t0 * HD,      HD);
  const v16h aq1 = load_frag(q + (size_t)t0 * HD + 32, HD);

  v8f o0 = {}, o1 = {}, o2 = {}, o3 = {};
  float rm[8], rl[8];
  #pragma unroll
  for (int r = 0; r < 8; r++) { rm[r] = -1e30f; rl[r] = 0.f; }

  for (int s = 0; s < L; s += 32) {
    // ---- S = Q K^T for 16 x 32 block of scores ----
    v16h kb0 = load_frag(k + (size_t)s * HD,            HD);
    v16h kb1 = load_frag(k + (size_t)s * HD + 32,       HD);
    v16h kc0 = load_frag(k + (size_t)(s + 16) * HD,     HD);
    v16h kc1 = load_frag(k + (size_t)(s + 16) * HD + 32, HD);
    v8f sa = {}, sb = {};
    sa = wmma16(aq0, kb0, sa); sa = wmma16(aq1, kb1, sa);
    sb = wmma16(aq0, kc0, sb); sb = wmma16(aq1, kc1, sb);

    // ---- online softmax over the 32 new columns ----
    float alpha[8];
    #pragma unroll
    for (int r = 0; r < 8; r++) {
      float mx = fmaxf(sa[r], sb[r]);
      #pragma unroll
      for (int off = 8; off >= 1; off >>= 1)
        mx = fmaxf(mx, __shfl_xor(mx, off, 32));   // reduce over 16-lane half
      float mnew = fmaxf(rm[r], mx);
      float a_   = __expf(rm[r] - mnew);
      float pa   = __expf(sa[r] - mnew);
      float pb   = __expf(sb[r] - mnew);
      float psum = pa + pb;
      #pragma unroll
      for (int off = 8; off >= 1; off >>= 1)
        psum += __shfl_xor(psum, off, 32);
      rl[r] = rl[r] * a_ + psum;
      rm[r] = mnew;
      alpha[r] = a_;
      int m = r + half * 8;
      pbuf[wave][m * 32 + nloc]      = (_Float16)pa;
      pbuf[wave][m * 32 + 16 + nloc] = (_Float16)pb;
    }
    #pragma unroll
    for (int r = 0; r < 8; r++) {
      o0[r] *= alpha[r]; o1[r] *= alpha[r];
      o2[r] *= alpha[r]; o3[r] *= alpha[r];
    }

    // ---- C-layout P -> A-fragment via LDS ----
    __syncthreads();
    v16h pf = load_frag(&pbuf[wave][0], 32);
    __syncthreads();

    // ---- O += P * V  (V is c-major: Bt[c,s]) ----
    v16h vb0 = load_frag(v + (size_t)0  * L + s, L);
    v16h vb1 = load_frag(v + (size_t)16 * L + s, L);
    v16h vb2 = load_frag(v + (size_t)32 * L + s, L);
    v16h vb3 = load_frag(v + (size_t)48 * L + s, L);
    o0 = wmma16(pf, vb0, o0);
    o1 = wmma16(pf, vb1, o1);
    o2 = wmma16(pf, vb2, o2);
    o3 = wmma16(pf, vb3, o3);
  }

  // ---- normalize and store t-major into a^T (b, L, C) ----
  #pragma unroll
  for (int r = 0; r < 8; r++) {
    float inv = 1.f / rl[r];
    int t = t0 + r + half * 8;
    size_t rowbase = ((size_t)b * L + t) * C + h * HD;
    at[rowbase + 0  + nloc] = (_Float16)(o0[r] * inv);
    at[rowbase + 16 + nloc] = (_Float16)(o1[r] * inv);
    at[rowbase + 32 + nloc] = (_Float16)(o2[r] * inv);
    at[rowbase + 48 + nloc] = (_Float16)(o3[r] * inv);
  }
}

// ---------------------------------------------------------------------------
// Proj GEMM + bias + residual: out(512x1024) = Pw(512x512) * a(512x1024)
// NT form with Bt = a^T (LxC).  fp32 output.
// ---------------------------------------------------------------------------
__global__ void proj_gemm_kernel(const _Float16* __restrict__ Wh,
                                 const _Float16* __restrict__ at,
                                 const float* __restrict__ pb,
                                 const float* __restrict__ x,
                                 float* __restrict__ out) {
  const int C = 512, L = 1024;
  const int b    = blockIdx.z;
  const int wave = threadIdx.x >> 5, lane = threadIdx.x & 31;
  const int half = lane >> 4,        nloc = lane & 15;
  const int m0 = blockIdx.x * 128 + (wave & 3) * 32;
  const int n0 = blockIdx.y * 64  + (wave >> 2) * 32;
  const _Float16* Bt = at + (size_t)b * L * C;

  v8f c00 = {}, c01 = {}, c10 = {}, c11 = {};
  for (int k = 0; k < C; k += 32) {
    __builtin_prefetch(Wh + (size_t)m0 * C + k + 32, 0, 1);
    __builtin_prefetch(Bt + (size_t)n0 * C + k + 32, 0, 1);
    v16h a0 = load_frag(Wh + (size_t)m0 * C + k, C);
    v16h a1 = load_frag(Wh + (size_t)(m0 + 16) * C + k, C);
    v16h b0 = load_frag(Bt + (size_t)n0 * C + k, C);
    v16h b1 = load_frag(Bt + (size_t)(n0 + 16) * C + k, C);
    c00 = wmma16(a0, b0, c00);
    c01 = wmma16(a0, b1, c01);
    c10 = wmma16(a1, b0, c10);
    c11 = wmma16(a1, b1, c11);
  }

  auto store_tile = [&](v8f acc, int tm, int tn) {
    #pragma unroll
    for (int r = 0; r < 8; r++) {
      int o = m0 + tm * 16 + r + half * 8;
      int l = n0 + tn * 16 + nloc;
      size_t idx = ((size_t)b * C + o) * L + l;
      out[idx] = acc[r] + pb[o] + x[idx];
    }
  };
  store_tile(c00, 0, 0); store_tile(c01, 0, 1);
  store_tile(c10, 1, 0); store_tile(c11, 1, 1);
}

// ---------------------------------------------------------------------------
extern "C" void kernel_launch(void* const* d_in, const int* in_sizes, int n_in,
                              void* d_out, int out_size, void* d_ws, size_t ws_size,
                              hipStream_t stream) {
  (void)in_sizes; (void)n_in; (void)out_size; (void)ws_size;
  const float* x      = (const float*)d_in[0];
  const float* norm_w = (const float*)d_in[1];
  const float* norm_b = (const float*)d_in[2];
  const float* qkv_w  = (const float*)d_in[3];
  const float* qkv_b  = (const float*)d_in[4];
  const float* proj_w = (const float*)d_in[5];
  const float* proj_b = (const float*)d_in[6];
  float* out = (float*)d_out;

  const size_t B = 8, C = 512, L = 1024, O3 = 1536, H = 8, HD = 64;

  uintptr_t base = (uintptr_t)d_ws;
  auto carve = [&](size_t nbytes) -> void* {
    base = (base + 255) & ~(uintptr_t)255;
    void* p = (void*)base;
    base += nbytes;
    return p;
  };
  _Float16* qkvw_h  = (_Float16*)carve(O3 * C * 2);            // 1.5 MB
  _Float16* projw_h = (_Float16*)carve(C * C * 2);             // 0.5 MB
  _Float16* xnt     = (_Float16*)carve(B * L * C * 2);         // 8 MB
  _Float16* qt      = (_Float16*)carve(B * H * L * HD * 2);    // 8 MB
  _Float16* kt      = (_Float16*)carve(B * H * L * HD * 2);    // 8 MB
  _Float16* vv      = (_Float16*)carve(B * H * HD * L * 2);    // 8 MB
  _Float16* at      = (_Float16*)carve(B * L * C * 2);         // 8 MB

  // 1) weight conversion fp32 -> f16
  cvt_f16_kernel<<<(int)((O3 * C + 255) / 256), 256, 0, stream>>>(qkv_w, qkvw_h,
                                                                  (int)(O3 * C));
  cvt_f16_kernel<<<(int)((C * C + 255) / 256), 256, 0, stream>>>(proj_w, projw_h,
                                                                 (int)(C * C));
  // 2) GroupNorm -> xn^T
  groupnorm_kernel<<<256, 256, 0, stream>>>(x, norm_w, norm_b, xnt);
  // 3) QKV GEMM + scatter
  qkv_gemm_kernel<<<dim3(12, 16, 8), 256, 0, stream>>>(qkvw_h, xnt, qkv_b,
                                                       qt, kt, vv);
  // 4) flash attention -> a^T
  attn_kernel<<<dim3(8, 64), 256, 0, stream>>>(qt, kt, vv, at);
  // 5) proj GEMM + bias + residual
  proj_gemm_kernel<<<dim3(4, 16, 8), 256, 0, stream>>>(projw_h, at, proj_b,
                                                       x, out);
}
